// SpatiallyVaryingDepthFilter_22694607192370
// MI455X (gfx1250) — matrix operand order repaired and checked by reading, not compiled, not run on hardware
//
#include <hip/hip_runtime.h>
#include <math.h>

typedef __attribute__((ext_vector_type(8)))  _Float16 v8h;
typedef __attribute__((ext_vector_type(16))) _Float16 v16h;
typedef __attribute__((ext_vector_type(8)))  float    v8f;

// Problem constants
#define BN   32
#define HH   128
#define WW   128
#define HW   (HH*WW)          // 16384
#define HP   130
#define WP   130
#define CIN  8                // guide(4) + filtered depth(4)
#define OD   320              // output channels
#define KTOT 72               // 8 * 9 taps
#define KPAD 96               // padded K (3 chunks of 32)

// Workspace layout
#define AW_OFF    0                                   // f16 [320][96]
#define YPAD_OFF  131072                              // f16 [B][130][130][8]
#define YPAD_HALVES ((size_t)BN*HP*WP*CIN)            // 4,326,400 halves
#define WS_NEEDED (YPAD_OFF + YPAD_HALVES*2)          // ~8.8 MB

// ---------------------------------------------------------------------------
// Kernel 0: zero ypad (borders must be zero; interior overwritten by kernel 2)
// ---------------------------------------------------------------------------
__global__ void svdf_zero(int4* __restrict__ p, int n16) {
    int i = blockIdx.x * blockDim.x + threadIdx.x;
    int stride = gridDim.x * blockDim.x;
    int4 z = make_int4(0, 0, 0, 0);
    for (; i < n16; i += stride) p[i] = z;
}

// ---------------------------------------------------------------------------
// Kernel 1: repack conv_w (320,8,3,3) fp32 -> Aw[o][k] f16, k = tap*8 + c,
//           zero-padded to KPAD.
// ---------------------------------------------------------------------------
__global__ void svdf_wprep(const float* __restrict__ conv_w,
                           _Float16* __restrict__ Aw) {
    int idx = blockIdx.x * blockDim.x + threadIdx.x;
    if (idx >= OD * KPAD) return;
    int k = idx % KPAD;
    int o = idx / KPAD;
    _Float16 v = (_Float16)0.0f;
    if (k < KTOT) {
        int tap = k >> 3;          // 0..8
        int c   = k & 7;           // input channel
        int dy  = tap / 3;
        int dx  = tap % 3;
        v = (_Float16)conv_w[((o * CIN + c) * 3 + dy) * 3 + dx];
    }
    Aw[idx] = v;
}

// ---------------------------------------------------------------------------
// Kernel 2: per-pixel guide->kernel logits, softmax over 9 taps per depth
//           channel, weighted 3x3 gather; write 8 f16 channels into ypad
//           interior at (h+1, w+1).
// ---------------------------------------------------------------------------
__global__ void svdf_filter(const float* __restrict__ in,
                            const float* __restrict__ wg_w,
                            const float* __restrict__ wg_b,
                            _Float16* __restrict__ ypad) {
    int idx = blockIdx.x * blockDim.x + threadIdx.x;
    if (idx >= BN * HW) return;
    int w = idx & (WW - 1);
    int h = (idx >> 7) & (HH - 1);
    int b = idx >> 14;

    const float* gb = in + (size_t)(b * 8) * HW + h * WW + w;
    float g0 = gb[0 * HW], g1 = gb[1 * HW], g2 = gb[2 * HW], g3 = gb[3 * HW];

    _Float16 yout[8];
    yout[0] = (_Float16)g0; yout[1] = (_Float16)g1;
    yout[2] = (_Float16)g2; yout[3] = (_Float16)g3;

    #pragma unroll
    for (int c = 0; c < 4; ++c) {
        float e[9];
        float m = -3.0e38f;
        #pragma unroll
        for (int t = 0; t < 9; ++t) {
            int o = c * 9 + t;
            float l = wg_b[o] + g0 * wg_w[o * 4 + 0] + g1 * wg_w[o * 4 + 1]
                              + g2 * wg_w[o * 4 + 2] + g3 * wg_w[o * 4 + 3];
            e[t] = l;
            m = fmaxf(m, l);
        }
        float s = 0.0f;
        #pragma unroll
        for (int t = 0; t < 9; ++t) { e[t] = __expf(e[t] - m); s += e[t]; }
        float inv = 1.0f / s;

        const float* db = in + (size_t)(b * 8 + 4 + c) * HW;
        float acc = 0.0f;
        #pragma unroll
        for (int t = 0; t < 9; ++t) {
            int dy = t / 3 - 1, dx = t % 3 - 1;
            int hh = h + dy, ww2 = w + dx;
            float v = 0.0f;
            if (hh >= 0 && hh < HH && ww2 >= 0 && ww2 < WW)
                v = db[hh * WW + ww2];
            acc += v * e[t] * inv;
        }
        yout[4 + c] = (_Float16)acc;
    }

    size_t o = (((size_t)b * HP + (h + 1)) * WP + (w + 1)) * CIN;
    #pragma unroll
    for (int c = 0; c < CIN; ++c) ypad[o + c] = yout[c];
}

// ---------------------------------------------------------------------------
// Kernel 3: implicit-GEMM 3x3 conv via WMMA.
//   M = 320 output channels (20 tiles of 16), split into 4 groups of 5 tiles.
//   N = pixel tiles of 16 consecutive w (32768 tiles).
//   K = 72 padded to 96 (3 chunks of 32), k = tap*8 + c.
// One wave per (mgroup, ntile): loads 3 B-fragments once, reuses them for
// 5 M-tiles -> 15 v_wmma_f32_16x16x32_f16 per wave, accum seeded with bias.
// ---------------------------------------------------------------------------
__global__ void __launch_bounds__(256)
svdf_gemm(const _Float16* __restrict__ Aw,
          const _Float16* __restrict__ ypad,
          const float* __restrict__ conv_b,
          float* __restrict__ out) {
    const int lane  = threadIdx.x & 31;
    const int wib   = threadIdx.x >> 5;
    const int gwave = blockIdx.x * 8 + wib;

    const int ntile  = gwave & 32767;   // pixel tile
    const int mgroup = gwave >> 15;     // 0..3 (5 M-tiles each)

    const int n  = lane & 15;           // N within tile / M-row for A
    const int hi = lane >> 4;           // K-half selector

    const int p  = ntile << 4;
    const int b  = p >> 14;
    const int q  = p & (HW - 1);
    const int h  = q >> 7;
    const int w0 = q & (WW - 1);        // multiple of 16

    // ---- B fragments: lane holds K = ck*32 + hi*8 + [0..7] and ck*32 + 16 + hi*8 + [0..7]
    //      for pixel (b, h, w0+n). Each 8-K subchunk = 8 channels of one 3x3 tap =
    //      one contiguous 16-byte load from ypad.
    v16h bf[3];
    #pragma unroll
    for (int ck = 0; ck < 3; ++ck) {
        #pragma unroll
        for (int s = 0; s < 2; ++s) {
            int tap = ck * 4 + s * 2 + hi;      // 0..11; >=9 is K-padding
            int tc  = tap > 8 ? 8 : tap;        // clamp for safe address
            int dy  = tc / 3;
            int dx  = tc % 3;
            const v8h* ptr = (const v8h*)(ypad +
                ((((size_t)b * HP) + h + dy) * WP + (w0 + n + dx)) * CIN);
            v8h v = *ptr;
            v8h z = {};
            v = (tap < 9) ? v : z;
            #pragma unroll
            for (int i = 0; i < 8; ++i) bf[ck][s * 8 + i] = v[i];
        }
    }

    // ---- 5 M-tiles, reusing bf
    #pragma unroll
    for (int mt = 0; mt < 5; ++mt) {
        const int mtile = mgroup * 5 + mt;
        const int obase = mtile * 16;

        // C/D layout: VGPR r, lanes 0-15 -> M=r, lanes 16-31 -> M=r+8
        v8f acc = *(const v8f*)(conv_b + obase + hi * 8);

        // A layout: lane holds row M = lane&15; halves[0..7] = K ck*32+hi*8,
        // halves[8..15] = K ck*32+16+hi*8 (two contiguous 16B loads).
        const _Float16* arow = Aw + (size_t)(obase + n) * KPAD;
        #pragma unroll
        for (int ck = 0; ck < 3; ++ck) {
            v8h a0 = *(const v8h*)(arow + ck * 32 + hi * 8);
            v8h a1 = *(const v8h*)(arow + ck * 32 + 16 + hi * 8);
            v16h af;
            #pragma unroll
            for (int i = 0; i < 8; ++i) { af[i] = a0[i]; af[8 + i] = a1[i]; }
            acc = __builtin_amdgcn_wmma_f32_16x16x32_f16(
                false, af, false, bf[ck], (short)0, acc, false, false);
        }

        // Store: out[b][obase + hi*8 + r][h][w0+n]
        float* op = out + ((size_t)(b * OD + obase + hi * 8)) * HW
                        + h * WW + w0 + n;
        #pragma unroll
        for (int r = 0; r < 8; ++r) op[(size_t)r * HW] = acc[r];
    }
}

// ---------------------------------------------------------------------------
extern "C" void kernel_launch(void* const* d_in, const int* in_sizes, int n_in,
                              void* d_out, int out_size, void* d_ws, size_t ws_size,
                              hipStream_t stream) {
    (void)in_sizes; (void)n_in; (void)out_size;
    if (ws_size < WS_NEEDED) return;

    const float* in     = (const float*)d_in[0];
    const float* wg_w   = (const float*)d_in[1];
    const float* wg_b   = (const float*)d_in[2];
    const float* conv_w = (const float*)d_in[3];
    const float* conv_b = (const float*)d_in[4];
    float* out = (float*)d_out;

    _Float16* Aw   = (_Float16*)((char*)d_ws + AW_OFF);
    _Float16* ypad = (_Float16*)((char*)d_ws + YPAD_OFF);

    // 0) zero ypad (16B granules)
    int n16 = (int)(YPAD_HALVES * 2 / 16);
    svdf_zero<<<2048, 256, 0, stream>>>((int4*)ypad, n16);

    // 1) weight repack
    svdf_wprep<<<(OD * KPAD + 255) / 256, 256, 0, stream>>>(conv_w, Aw);

    // 2) spatially-varying filter + concat, f16 staging
    svdf_filter<<<(BN * HW + 255) / 256, 256, 0, stream>>>(in, wg_w, wg_b, ypad);

    // 3) WMMA implicit GEMM: 32768 ntiles * 4 mgroups = 131072 waves
    svdf_gemm<<<16384, 256, 0, stream>>>(Aw, ypad, conv_b, out);
}
